// SelfAttention_8263517077849
// MI455X (gfx1250) — matrix operand order repaired and checked by reading, not compiled, not run on hardware
//
#include <hip/hip_runtime.h>
#include <hip/hip_bf16.h>

typedef __bf16 bf16_t;
typedef __attribute__((ext_vector_type(16))) __bf16 bf16x16;
typedef __attribute__((ext_vector_type(8)))  __bf16 bf16x8;
typedef __attribute__((ext_vector_type(8)))  float  f32x8;

union BFrag { bf16x16 v; bf16x8 h[2]; };

// ---- WMMA wrapper: D = A(16x32 bf16) x B(32x16 bf16) + C(16x16 f32) ----
__device__ __forceinline__ f32x8 wmma_bf16(bf16x16 a, bf16x16 b, f32x8 c) {
    return __builtin_amdgcn_wmma_f32_16x16x32_bf16(
        /*neg_a=*/false, a, /*neg_b=*/false, b,
        /*c_mod=*/(short)0, c, /*reuse_a=*/false, /*reuse_b=*/false);
}

// A-matrix 16x32 bf16 fragment. Lane l<16: row l, K {0..7,16..23};
// lane l>=16: row l-16, K {8..15,24..31}. ldk multiple of 8 -> 16B aligned.
__device__ __forceinline__ bf16x16 ldA(const bf16_t* rowBase, int ldk, int lane) {
    int r = lane & 15, hh = lane >> 4;
    const bf16_t* p = rowBase + (size_t)r * ldk + hh * 8;
    BFrag f;
    f.h[0] = *(const bf16x8*)(p);
    f.h[1] = *(const bf16x8*)(p + 16);
    return f.v;
}

// B-matrix 32x16 bf16 fragment. Lane l<16: col l, K {0..7,16..23}; lane>=16: col l-16, K {8..15,24..31}.
__device__ __forceinline__ bf16x16 ldB(const bf16_t* colBase, int ldk, int lane) {
    int c = lane & 15, hh = lane >> 4;
    const bf16_t* p = colBase + (size_t)c * ldk + hh * 8;
    BFrag f;
    f.h[0] = *(const bf16x8*)(p);
    f.h[1] = *(const bf16x8*)(p + 16);
    return f.v;
}

// A fragment sourced from an fp32 LDS matrix, converted to bf16 on the fly.
__device__ __forceinline__ bf16x16 ldA_f32(const float* rowBase, int ld, int lane) {
    int r = lane & 15, hh = lane >> 4;
    const float* p = rowBase + (size_t)r * ld + hh * 8;
    bf16x16 out;
#pragma unroll
    for (int t = 0; t < 8; t++) out[t] = (bf16_t)p[t];
#pragma unroll
    for (int t = 0; t < 8; t++) out[8 + t] = (bf16_t)p[16 + t];
    return out;
}

// ---- Async global->LDS (ASYNCcnt-tracked), per-lane 16B transfers ----------
__device__ __forceinline__ void async_ld_b128(unsigned lds_off, const void* gaddr) {
    asm volatile("global_load_async_to_lds_b128 %0, %1, off"
                 :: "v"(lds_off), "v"(gaddr) : "memory");
}
template <int N>
__device__ __forceinline__ void wait_asynccnt() {
    asm volatile("s_wait_asynccnt %0" :: "n"(N) : "memory");
}

// ---------------- Kernel 0a: x (fp32 [b][c][n]) -> xbT (bf16 [b][n][c]) -----
__global__ __launch_bounds__(256) void x_to_bf16T_k(const float* __restrict__ x,
                                                    bf16_t* __restrict__ xbT) {
    __shared__ bf16_t tile[32][33];
    int b = blockIdx.z, c0 = blockIdx.y * 32, n0 = blockIdx.x * 32;
    for (int yy = threadIdx.y; yy < 32; yy += 8)
        tile[yy][threadIdx.x] =
            (bf16_t)x[((size_t)b * 512 + c0 + yy) * 4096 + n0 + threadIdx.x];
    __syncthreads();
    for (int yy = threadIdx.y; yy < 32; yy += 8)
        xbT[((size_t)b * 4096 + n0 + yy) * 512 + c0 + threadIdx.x] = tile[threadIdx.x][yy];
}

// ---------------- Kernel 0b: weights -> bf16 (theta|phi|g concat, and w_o) ---
__global__ __launch_bounds__(256) void conv_weights_k(
    const float* __restrict__ wt, const float* __restrict__ wp,
    const float* __restrict__ wg, const float* __restrict__ wo,
    bf16_t* __restrict__ Wcat, bf16_t* __restrict__ woB) {
    int i = blockIdx.x * 256 + threadIdx.x;
    const int nW = 384 * 512;
    if (i < nW) {
        int m = i >> 9, k = i & 511;
        float v = (m < 64) ? wt[m * 512 + k]
                 : (m < 128) ? wp[(m - 64) * 512 + k]
                             : wg[(m - 128) * 512 + k];
        Wcat[i] = (bf16_t)v;
    } else {
        int j = i - nW;
        if (j < 512 * 256) woB[j] = (bf16_t)wo[j];
    }
}

// ---------------- Kernel 1: fused projection GEMM ---------------------------
__global__ __launch_bounds__(256) void proj_gemm_k(
    const bf16_t* __restrict__ Wcat, const bf16_t* __restrict__ xbT,
    bf16_t* __restrict__ theta, bf16_t* __restrict__ phiF, bf16_t* __restrict__ gF) {
    const int lane = threadIdx.x & 31, wv = threadIdx.x >> 5;
    const int wm = wv & 1, wn = wv >> 1;
    const int b = blockIdx.z;
    const int m0 = blockIdx.y * 64 + wm * 32;
    const int n0 = blockIdx.x * 128 + wn * 32;
    const bf16_t* Xb = xbT + (size_t)b * 4096 * 512;
    f32x8 acc[2][2] = {};
#pragma unroll 4
    for (int k = 0; k < 512; k += 32) {
        bf16x16 a0 = ldA(Wcat + (size_t)m0 * 512 + k, 512, lane);
        bf16x16 a1 = ldA(Wcat + (size_t)(m0 + 16) * 512 + k, 512, lane);
        bf16x16 b0 = ldB(Xb + (size_t)n0 * 512 + k, 512, lane);
        bf16x16 b1 = ldB(Xb + (size_t)(n0 + 16) * 512 + k, 512, lane);
        acc[0][0] = wmma_bf16(a0, b0, acc[0][0]);
        acc[0][1] = wmma_bf16(a0, b1, acc[0][1]);
        acc[1][0] = wmma_bf16(a1, b0, acc[1][0]);
        acc[1][1] = wmma_bf16(a1, b1, acc[1][1]);
    }
    const int h = lane >> 4, cc = lane & 15;
#pragma unroll
    for (int i = 0; i < 2; i++)
#pragma unroll
        for (int j = 0; j < 2; j++)
#pragma unroll
            for (int r = 0; r < 8; r++) {
                int m = m0 + i * 16 + 8 * h + r;
                int n = n0 + j * 16 + cc;
                bf16_t bv = (bf16_t)acc[i][j][r];
                if (m < 64)       theta[((size_t)b * 64 + m) * 4096 + n] = bv;
                else if (m < 128) phiF[((size_t)b * 64 + (m - 64)) * 4096 + n] = bv;
                else              gF[((size_t)b * 256 + (m - 128)) * 4096 + n] = bv;
            }
}

// ---------------- Kernel 2: 2x2 maxpool; phi transposed to [k][d], g as [v][k]
__global__ __launch_bounds__(256) void pool_k(
    const bf16_t* __restrict__ phiF, const bf16_t* __restrict__ gF,
    bf16_t* __restrict__ phiT, bf16_t* __restrict__ gP) {
    int i = blockIdx.x * 256 + threadIdx.x;
    const int nPhi = 16 * 64 * 1024;
    if (i < nPhi) {
        int b = i / (64 * 1024), r = i % (64 * 1024), d = r >> 10, k = r & 1023;
        int hh = k >> 5, ww = k & 31;
        const bf16_t* s = phiF + ((size_t)b * 64 + d) * 4096;
        int n = hh * 128 + ww * 2;
        float m = fmaxf(fmaxf((float)s[n], (float)s[n + 1]),
                        fmaxf((float)s[n + 64], (float)s[n + 65]));
        phiT[((size_t)b * 1024 + k) * 64 + d] = (bf16_t)m;
    } else {
        int j = i - nPhi;
        if (j < 16 * 256 * 1024) {
            int b = j / (256 * 1024), r = j % (256 * 1024), v = r >> 10, k = r & 1023;
            int hh = k >> 5, ww = k & 31;
            const bf16_t* s = gF + ((size_t)b * 256 + v) * 4096;
            int n = hh * 128 + ww * 2;
            float m = fmaxf(fmaxf((float)s[n], (float)s[n + 1]),
                            fmaxf((float)s[n + 64], (float)s[n + 65]));
            gP[((size_t)b * 256 + v) * 1024 + k] = (bf16_t)m;
        }
    }
}

// ---------------- Kernel 3: attention (full-row softmax in LDS) -------------
// LDS: S 64xSLD f32 (264KB) | Q 64xQLD bf16 (9KB) | red 512 f32 (2KB)
//      | per-wave double-buffered async g staging (8 waves x 2 x 2560B = 40KB)
#define SLD 1032
#define QLD 72
#define GROWB 80                       // padded LDS row stride (bytes) for g tiles
#define GBUFB (32 * GROWB)             // one staging buffer = 2560 B
#define LDS_GBASE (64 * SLD * 4 + 64 * QLD * 2 + 512 * 4)  // 275456
__global__ __launch_bounds__(256) void attn_k(
    const bf16_t* __restrict__ theta, const bf16_t* __restrict__ phiT,
    const bf16_t* __restrict__ gP, bf16_t* __restrict__ O) {
    extern __shared__ char smem[];
    float* S = (float*)smem;
    bf16_t* Q = (bf16_t*)(smem + 64 * SLD * 4);
    float* red = (float*)(smem + 64 * SLD * 4 + 64 * QLD * 2);
    const int tid = threadIdx.x, lane = tid & 31, wv = tid >> 5;
    const int b = blockIdx.y, n0 = blockIdx.x * 64;

    // Stage Q^T: ldsQ[n][d] from theta[b][d][n0..n0+63]
    const bf16_t* th = theta + (size_t)b * 64 * 4096;
    for (int i = tid; i < 64 * 64; i += 256) {
        int d = i >> 6, j = i & 63;
        Q[j * QLD + d] = th[(size_t)d * 4096 + n0 + j];
    }
    __syncthreads();

    // Phase 1: S[n][k] = sum_d Q[n][d] * phiT[k][d]; wave owns 128 keys.
    const bf16_t* ph = phiT + (size_t)b * 1024 * 64;
    const int h = lane >> 4, cc = lane & 15;
    for (int kb = 0; kb < 128; kb += 32) {
        int kcol = wv * 128 + kb;
        f32x8 acc[4][2] = {};
#pragma unroll
        for (int d = 0; d < 64; d += 32) {
            bf16x16 b0 = ldB(ph + (size_t)kcol * 64 + d, 64, lane);
            bf16x16 b1 = ldB(ph + (size_t)(kcol + 16) * 64 + d, 64, lane);
#pragma unroll
            for (int ni = 0; ni < 4; ni++) {
                bf16x16 aa = ldA(Q + (size_t)(ni * 16) * QLD + d, QLD, lane);
                acc[ni][0] = wmma_bf16(aa, b0, acc[ni][0]);
                acc[ni][1] = wmma_bf16(aa, b1, acc[ni][1]);
            }
        }
#pragma unroll
        for (int ni = 0; ni < 4; ni++)
#pragma unroll
            for (int kj = 0; kj < 2; kj++)
#pragma unroll
                for (int r = 0; r < 8; r++)
                    S[(ni * 16 + 8 * h + r) * SLD + kcol + kj * 16 + cc] = acc[ni][kj][r];
    }
    __syncthreads();

    // Softmax over k (1024) per row: 4 threads per row, 256 elems (64 float4) each.
    {
        int row = tid >> 2, part = tid & 3;
        float* Sr = S + (size_t)row * SLD + part * 256;
        float4* Sr4 = (float4*)Sr;
        float mx = -1e30f;
        for (int k = 0; k < 64; k++) {
            float4 v = Sr4[k];
            mx = fmaxf(mx, fmaxf(fmaxf(v.x, v.y), fmaxf(v.z, v.w)));
        }
        red[row * 4 + part] = mx;
        __syncthreads();
        float rmx = fmaxf(fmaxf(red[row * 4 + 0], red[row * 4 + 1]),
                          fmaxf(red[row * 4 + 2], red[row * 4 + 3]));
        float s = 0.f;
        for (int k = 0; k < 64; k++) {
            float4 v = Sr4[k];
            v.x = __expf(v.x - rmx); v.y = __expf(v.y - rmx);
            v.z = __expf(v.z - rmx); v.w = __expf(v.w - rmx);
            Sr4[k] = v;
            s += (v.x + v.y) + (v.z + v.w);
        }
        red[256 + row * 4 + part] = s;
        __syncthreads();
        float inv = 1.0f / (red[256 + row * 4 + 0] + red[256 + row * 4 + 1] +
                            red[256 + row * 4 + 2] + red[256 + row * 4 + 3]);
        for (int k = 0; k < 64; k++) {
            float4 v = Sr4[k];
            v.x *= inv; v.y *= inv; v.z *= inv; v.w *= inv;
            Sr4[k] = v;
        }
    }
    __syncthreads();

    // Phase 2: O[n][v] = sum_k S[n][k] * g[v][k]; wave owns 32 v-columns.
    // g tiles (32 rows x 64B) streamed via double-buffered async global->LDS;
    // per-wave private staging => only s_wait_asynccnt, no barriers.
    const bf16_t* gb = gP + (size_t)b * 256 * 1024;
    const unsigned ldsWave = (unsigned)LDS_GBASE + (unsigned)wv * (2 * GBUFB);
    const int rr = lane >> 2, rc = lane & 3;  // lane -> (row, 16B-chunk) for issue
    auto issue_block = [&](unsigned buf, int kb) {
#pragma unroll
        for (int i = 0; i < 4; i++) {
            int row = i * 8 + rr;  // 0..31
            const char* ga = (const char*)(gb + ((size_t)(wv * 32 + row)) * 1024 + kb) + rc * 16;
            unsigned lo = ldsWave + buf * GBUFB + (unsigned)row * GROWB + (unsigned)rc * 16;
            async_ld_b128(lo, ga);
        }
    };
    f32x8 oacc[4][2] = {};
    issue_block(0, 0);
    for (int kbi = 0; kbi < 32; kbi++) {
        int kb = kbi * 32;
        if (kbi + 1 < 32) {
            issue_block((kbi + 1) & 1, kb + 32);
            wait_asynccnt<4>();   // current buffer's 4 transfers retired (in-order)
        } else {
            wait_asynccnt<0>();
        }
        const bf16_t* wbuf = (const bf16_t*)(smem + ldsWave + (kbi & 1) * GBUFB);
        bf16x16 b0 = ldB(wbuf, GROWB / 2, lane);
        bf16x16 b1 = ldB(wbuf + 16 * (GROWB / 2), GROWB / 2, lane);
#pragma unroll
        for (int ni = 0; ni < 4; ni++) {
            bf16x16 aa = ldA_f32(S + (size_t)(ni * 16) * SLD + kb, SLD, lane);
            oacc[ni][0] = wmma_bf16(aa, b0, oacc[ni][0]);
            oacc[ni][1] = wmma_bf16(aa, b1, oacc[ni][1]);
        }
    }
#pragma unroll
    for (int ni = 0; ni < 4; ni++)
#pragma unroll
        for (int vj = 0; vj < 2; vj++)
#pragma unroll
            for (int r = 0; r < 8; r++) {
                int n = n0 + ni * 16 + 8 * h + r;
                int v = wv * 32 + vj * 16 + cc;
                O[((size_t)b * 4096 + n) * 256 + v] = (bf16_t)oacc[ni][vj][r];
            }
}

// ---------------- Kernel 4: output GEMM + residual epilogue -----------------
__global__ __launch_bounds__(256) void out_gemm_k(
    const bf16_t* __restrict__ woB, const bf16_t* __restrict__ O,
    const float* __restrict__ x, const float* __restrict__ gammaP,
    float* __restrict__ out) {
    const int lane = threadIdx.x & 31, wv = threadIdx.x >> 5;
    const int wm = wv & 1, wn = wv >> 1;
    const int b = blockIdx.z;
    const int m0 = blockIdx.y * 64 + wm * 32;
    const int n0 = blockIdx.x * 128 + wn * 32;
    const bf16_t* Ob = O + (size_t)b * 4096 * 256;
    f32x8 acc[2][2] = {};
#pragma unroll 4
    for (int k = 0; k < 256; k += 32) {
        bf16x16 a0 = ldA(woB + (size_t)m0 * 256 + k, 256, lane);
        bf16x16 a1 = ldA(woB + (size_t)(m0 + 16) * 256 + k, 256, lane);
        bf16x16 b0 = ldB(Ob + (size_t)n0 * 256 + k, 256, lane);
        bf16x16 b1 = ldB(Ob + (size_t)(n0 + 16) * 256 + k, 256, lane);
        acc[0][0] = wmma_bf16(a0, b0, acc[0][0]);
        acc[0][1] = wmma_bf16(a0, b1, acc[0][1]);
        acc[1][0] = wmma_bf16(a1, b0, acc[1][0]);
        acc[1][1] = wmma_bf16(a1, b1, acc[1][1]);
    }
    const float gamma = *gammaP;
    const int h = lane >> 4, cc = lane & 15;
#pragma unroll
    for (int i = 0; i < 2; i++)
#pragma unroll
        for (int j = 0; j < 2; j++)
#pragma unroll
            for (int r = 0; r < 8; r++) {
                int m = m0 + i * 16 + 8 * h + r;
                int n = n0 + j * 16 + cc;
                size_t idx = ((size_t)b * 512 + m) * 4096 + n;
                out[idx] = gamma * acc[i][j][r] + x[idx];
            }
}

extern "C" void kernel_launch(void* const* d_in, const int* in_sizes, int n_in,
                              void* d_out, int out_size, void* d_ws, size_t ws_size,
                              hipStream_t stream) {
    const float* x  = (const float*)d_in[0];
    const float* wt = (const float*)d_in[1];
    const float* wp = (const float*)d_in[2];
    const float* wg = (const float*)d_in[3];
    const float* wo = (const float*)d_in[4];
    const float* gm = (const float*)d_in[5];
    float* out = (float*)d_out;

    char* ws = (char*)d_ws;
    size_t off = 0;
    bf16_t* xbT  = (bf16_t*)(ws + off); off += (size_t)16 * 4096 * 512 * 2;  // 64 MB
    bf16_t* Wcat = (bf16_t*)(ws + off); off += (size_t)384 * 512 * 2;
    bf16_t* woB  = (bf16_t*)(ws + off); off += (size_t)512 * 256 * 2;
    bf16_t* thB  = (bf16_t*)(ws + off); off += (size_t)16 * 64 * 4096 * 2;   // 8 MB
    bf16_t* phiF = (bf16_t*)(ws + off); off += (size_t)16 * 64 * 4096 * 2;   // 8 MB
    bf16_t* gF   = (bf16_t*)(ws + off); off += (size_t)16 * 256 * 4096 * 2;  // 32 MB
    bf16_t* phiT = (bf16_t*)(ws + off); off += (size_t)16 * 1024 * 64 * 2;   // 2 MB
    bf16_t* gP   = (bf16_t*)(ws + off); off += (size_t)16 * 256 * 1024 * 2;  // 8 MB
    bf16_t* Obuf = (bf16_t*)(ws + off); off += (size_t)16 * 4096 * 256 * 2;  // 32 MB

    x_to_bf16T_k<<<dim3(128, 16, 16), dim3(32, 8), 0, stream>>>(x, xbT);
    conv_weights_k<<<1280, 256, 0, stream>>>(wt, wp, wg, wo, Wcat, woB);
    proj_gemm_k<<<dim3(32, 6, 16), 256, 0, stream>>>(Wcat, xbT, thB, phiF, gF);
    pool_k<<<20480, 256, 0, stream>>>(phiF, gF, phiT, gP);
    size_t attnLds = (size_t)LDS_GBASE + 8 * 2 * GBUFB;  // 316,416 B
    attn_k<<<dim3(64, 16), 256, attnLds, stream>>>(thB, phiT, gP, Obuf);
    out_gemm_k<<<dim3(32, 8, 16), 256, 0, stream>>>(woB, Obuf, x, gm, out);
}
